// HybridGNN_51737176048170
// MI455X (gfx1250) — matrix operand order repaired and checked by reading, not compiled, not run on hardware
//
#include <hip/hip_runtime.h>
#include <hip/hip_bf16.h>

// ---------------------------------------------------------------------------
// HybridGNN for MI455X (gfx1250, wave32). Heavy GEMMs use v_wmma_f32_16x16x32_bf16.
// Activations live in bf16 between layers, padded to 64-row multiples so the
// node GEMM runs guard-free (no exec-mask churn around WMMAs). WMMA B operands
// are pre-packed into the ISA lane layout; one wave amortizes each B fragment
// over 4 (node GEMM) or 2 (edge kernel) M-tiles.
// ---------------------------------------------------------------------------

#define NBATCH 64

typedef __attribute__((ext_vector_type(16))) __bf16 v16bf;
typedef __attribute__((ext_vector_type(8)))  float  v8f;

union FragAB { v16bf v; uint4 q[2]; unsigned short s[16]; };

__device__ __forceinline__ unsigned short f32_to_bf16(float f) {
  union { float f; unsigned int u; } x; x.f = f;
  unsigned int r = x.u + 0x7FFFu + ((x.u >> 16) & 1u);   // round-to-nearest-even
  return (unsigned short)(r >> 16);
}
__device__ __forceinline__ float bf16_to_f32(unsigned short u) {
  union { unsigned int u; float f; } x; x.u = (unsigned int)u << 16;
  return x.f;
}

// ---- pack f32 weight [K, Nout] into WMMA B-fragment order -----------------
// P[((kt*NT + nt)*32 + lane)*16 + e] = bf16( W[(kt*32 + (e>>3)*16 + (lane>>4)*8 + (e&7))*Nout
//                                              + nt*16 + (lane&15)] )
__global__ void pack_w_bf16(const float* __restrict__ W, unsigned short* __restrict__ P,
                            int K, int Nout) {
  int idx = blockIdx.x * blockDim.x + threadIdx.x;
  if (idx >= K * Nout) return;
  int e    = idx & 15;
  int lane = (idx >> 4) & 31;
  int t    = idx >> 9;                  // kt*NT + nt
  int NT   = Nout >> 4;
  int nt   = t % NT, kt = t / NT;
  int krow = kt * 32 + (e >> 3) * 16 + (lane >> 4) * 8 + (e & 7);
  int ncol = nt * 16 + (lane & 15);
  P[idx] = f32_to_bf16(W[(size_t)krow * Nout + ncol]);
}

// ---- one-time activation convert f32 -> bf16 ------------------------------
__global__ void cvt_f32_bf16(const float* __restrict__ x, unsigned short* __restrict__ o, int n) {
  int i = (blockIdx.x * blockDim.x + threadIdx.x) * 4;
  if (i >= n) return;
  float4 v = *(const float4*)(x + i);
  ushort4 u;
  u.x = f32_to_bf16(v.x); u.y = f32_to_bf16(v.y);
  u.z = f32_to_bf16(v.z); u.w = f32_to_bf16(v.w);
  *(ushort4*)(o + i) = u;
}

__global__ void zero_u16(unsigned short* p, int n) {
  int i = blockIdx.x * blockDim.x + threadIdx.x;
  if (i < n) p[i] = 0;
}
__global__ void zero_f32(float* p, int n) {
  int i = blockIdx.x * blockDim.x + threadIdx.x;
  if (i < n) p[i] = 0.f;
}

#define WMMA_BF16(a, b, c) \
  __builtin_amdgcn_wmma_f32_16x16x32_bf16(false, (a).v, false, (b).v, (short)0, (c), false, false)

// ---- node GEMM: out_bf16[M64,Nout] = A_bf16[M64,K] @ Wpacked ---------------
// M64 is a multiple of 64. One wave: 4 M-tiles x 1 N-tile, completely
// guard-free. Loads grouped so 4 independent WMMAs issue back-to-back.
__global__ void gemm_xw_bf16(const unsigned short* __restrict__ A,
                             const unsigned short* __restrict__ Wp,
                             unsigned short* __restrict__ outb,
                             int M64, int K, int Nout) {
  int w = threadIdx.x >> 5, lid = threadIdx.x & 31;
  int NT = Nout >> 4;
  int tile = blockIdx.x * 8 + w;
  if (tile >= (M64 >> 6) * NT) return;
  int mt0 = (tile / NT) * 4, nt = tile % NT;
  int half = lid >> 4, lc = lid & 15;

  const unsigned short* a0 = A + (size_t)(mt0 * 16 + lc) * K;
  const unsigned short* a1 = a0 + (size_t)16 * K;
  const unsigned short* a2 = a1 + (size_t)16 * K;
  const unsigned short* a3 = a2 + (size_t)16 * K;

  v8f c0 = {}, c1 = {}, c2 = {}, c3 = {};
  for (int kt = 0; kt < (K >> 5); ++kt) {
    int k0 = kt * 32 + half * 8;
    FragAB b, f0, f1, f2, f3;
    const unsigned short* bp = Wp + ((((size_t)kt * NT + nt) * 32 + lid) << 4);
    b.q[0]  = *(const uint4*)bp;           b.q[1]  = *(const uint4*)(bp + 8);
    f0.q[0] = *(const uint4*)(a0 + k0);    f0.q[1] = *(const uint4*)(a0 + k0 + 16);
    f1.q[0] = *(const uint4*)(a1 + k0);    f1.q[1] = *(const uint4*)(a1 + k0 + 16);
    f2.q[0] = *(const uint4*)(a2 + k0);    f2.q[1] = *(const uint4*)(a2 + k0 + 16);
    f3.q[0] = *(const uint4*)(a3 + k0);    f3.q[1] = *(const uint4*)(a3 + k0 + 16);
    __builtin_prefetch(bp + (size_t)NT * 512, 0, 1);  // next K-tile of packed W
    c0 = WMMA_BF16(f0, b, c0);
    c1 = WMMA_BF16(f1, b, c1);
    c2 = WMMA_BF16(f2, b, c2);
    c3 = WMMA_BF16(f3, b, c3);
  }
  int col = nt * 16 + lc;
  unsigned short* o0 = outb + (size_t)(mt0 * 16 + half * 8) * Nout + col;
#pragma unroll
  for (int r = 0; r < 8; ++r) o0[(size_t)r * Nout] = f32_to_bf16(c0[r]);
  o0 += (size_t)16 * Nout;
#pragma unroll
  for (int r = 0; r < 8; ++r) o0[(size_t)r * Nout] = f32_to_bf16(c1[r]);
  o0 += (size_t)16 * Nout;
#pragma unroll
  for (int r = 0; r < 8; ++r) o0[(size_t)r * Nout] = f32_to_bf16(c2[r]);
  o0 += (size_t)16 * Nout;
#pragma unroll
  for (int r = 0; r < 8; ++r) o0[(size_t)r * Nout] = f32_to_bf16(c3[r]);
}

// ---- fused edge kernel: 32 edges/WG, 8 waves x 2 N-tiles x 2 M-tiles ------
__global__ void edge_msg_scatter(const unsigned short* __restrict__ xtb,   // [N,256] bf16
                                 const int* __restrict__ eidx,             // [2,E]
                                 const float* __restrict__ eattr,          // [E,64]
                                 const unsigned short* __restrict__ msgWp, // packed 512x256
                                 const float* __restrict__ msgB,           // [256]
                                 const unsigned short* __restrict__ WeP,   // packed 64x256
                                 float* __restrict__ agg, int E_) {
  __shared__ __align__(16) unsigned short sA[32 * 512];  // [x_i | x_j] bf16
  __shared__ __align__(16) unsigned short sE[32 * 64];   // edge_attr bf16
  __shared__ int sDst[32];
  int t = threadIdx.x;
  int ebase = blockIdx.x * 32;
  int r = t >> 3;                                        // 32 rows, 8 threads/row
  int er = ebase + r; if (er >= E_) er = E_ - 1;
  if (t < 32) { int e2 = ebase + t; if (e2 >= E_) e2 = E_ - 1; sDst[t] = eidx[(size_t)E_ + e2]; }
  int src = eidx[er];
  int dst = eidx[(size_t)E_ + er];
  // stage concat row: cols 0..255 = xt[dst] (x_i), 256..511 = xt[src] (x_j)
  int cseg = (t & 7) * 64;                               // 64 bf16 = 8 uint4
  const unsigned short* xr = (cseg < 256) ? (xtb + (size_t)dst * 256 + cseg)
                                          : (xtb + (size_t)src * 256 + (cseg - 256));
  uint4* d4 = (uint4*)(sA + r * 512 + cseg);
  const uint4* s4 = (const uint4*)xr;
#pragma unroll
  for (int i = 0; i < 8; ++i) d4[i] = s4[i];
  {
    int rc = (t & 7) * 8;
    const float* ea = eattr + (size_t)er * 64 + rc;
    unsigned short* ed = sE + r * 64 + rc;
#pragma unroll
    for (int i = 0; i < 8; ++i) ed[i] = f32_to_bf16(ea[i]);
  }
  __syncthreads();
  int w = t >> 5, lid = t & 31, half = lid >> 4, lc = lid & 15;
  bool full = (ebase + 32 <= E_);                        // scalar (blockIdx-derived)
  for (int sub = 0; sub < 2; ++sub) {
    int nt = w * 2 + sub;
    v8f c0 = {}, c1 = {}, ce0 = {}, ce1 = {};
    for (int kt = 0; kt < 16; ++kt) {                    // msg GEMM, K=512
      FragAB a0, a1, b;
      const unsigned short* bp = msgWp + ((((kt << 4) + nt) * 32 + lid) << 4);
      b.q[0] = *(const uint4*)bp;         b.q[1] = *(const uint4*)(bp + 8);
      const unsigned short* ar = sA + lc * 512 + kt * 32 + half * 8;
      a0.q[0] = *(const uint4*)ar;        a0.q[1] = *(const uint4*)(ar + 16);
      a1.q[0] = *(const uint4*)(ar + 16 * 512);
      a1.q[1] = *(const uint4*)(ar + 16 * 512 + 16);
      __builtin_prefetch(bp + 8192, 0, 1);
      c0 = WMMA_BF16(a0, b, c0);
      c1 = WMMA_BF16(a1, b, c1);
    }
    for (int kt = 0; kt < 2; ++kt) {                     // et GEMM, K=64
      FragAB a0, a1, b;
      const unsigned short* bp = WeP + ((((kt << 4) + nt) * 32 + lid) << 4);
      b.q[0] = *(const uint4*)bp;         b.q[1] = *(const uint4*)(bp + 8);
      const unsigned short* ar = sE + lc * 64 + kt * 32 + half * 8;
      a0.q[0] = *(const uint4*)ar;        a0.q[1] = *(const uint4*)(ar + 16);
      a1.q[0] = *(const uint4*)(ar + 16 * 64);
      a1.q[1] = *(const uint4*)(ar + 16 * 64 + 16);
      ce0 = WMMA_BF16(a0, b, ce0);
      ce1 = WMMA_BF16(a1, b, ce1);
    }
    int col = nt * 16 + lc;
    float bias = msgB[col];
    if (full) {                                          // no per-lane guards
#pragma unroll
      for (int rr = 0; rr < 8; ++rr) {
        int erow = rr + half * 8;
        atomicAdd(&agg[(size_t)sDst[erow] * 256 + col], (c0[rr] + bias) * ce0[rr]);
        atomicAdd(&agg[(size_t)sDst[16 + erow] * 256 + col], (c1[rr] + bias) * ce1[rr]);
      }
    } else {
#pragma unroll
      for (int rr = 0; rr < 8; ++rr) {
        int erow = rr + half * 8;
        if (ebase + erow < E_)
          atomicAdd(&agg[(size_t)sDst[erow] * 256 + col], (c0[rr] + bias) * ce0[rr]);
        if (ebase + 16 + erow < E_)
          atomicAdd(&agg[(size_t)sDst[16 + erow] * 256 + col], (c1[rr] + bias) * ce1[rr]);
      }
    }
  }
}

// ---- GConv epilogue: LayerNorm + ReLU, writes bf16 activations ------------
__global__ void ln_relu_bf16(const float* __restrict__ x, const float* __restrict__ g,
                             const float* __restrict__ b, unsigned short* __restrict__ out,
                             int M) {
  int w = threadIdx.x >> 5, lid = threadIdx.x & 31;
  int row = blockIdx.x * 8 + w;
  if (row >= M) return;
  const float* xr = x + (size_t)row * 256;
  float v[8]; float s = 0.f, ss = 0.f;
#pragma unroll
  for (int i = 0; i < 8; ++i) {
    float tv = xr[lid + i * 32];
    v[i] = tv; s += tv; ss += tv * tv;
  }
#pragma unroll
  for (int o = 16; o > 0; o >>= 1) { s += __shfl_xor(s, o, 32); ss += __shfl_xor(ss, o, 32); }
  float mean = s * (1.f / 256.f);
  float var  = ss * (1.f / 256.f) - mean * mean;
  float inv  = rsqrtf(var + 1e-5f);
  unsigned short* orow = out + (size_t)row * 256;
#pragma unroll
  for (int i = 0; i < 8; ++i) {
    int cidx = lid + i * 32;
    float y = (v[i] - mean) * inv * g[cidx] + b[cidx];
    orow[cidx] = f32_to_bf16(fmaxf(y, 0.f));
  }
}

// ---- f32 LayerNorm (+residual) for the MHA tail ---------------------------
__global__ void ln_kernel(const float* __restrict__ x, const float* __restrict__ res,
                          const float* __restrict__ g, const float* __restrict__ b,
                          float* __restrict__ out, int M) {
  int w = threadIdx.x >> 5, lid = threadIdx.x & 31;
  int row = blockIdx.x * 8 + w;
  if (row >= M) return;
  const float* xr = x + (size_t)row * 256;
  float v[8]; float s = 0.f, ss = 0.f;
#pragma unroll
  for (int i = 0; i < 8; ++i) {
    float tv = xr[lid + i * 32] + res[(size_t)row * 256 + lid + i * 32];
    v[i] = tv; s += tv; ss += tv * tv;
  }
#pragma unroll
  for (int o = 16; o > 0; o >>= 1) { s += __shfl_xor(s, o, 32); ss += __shfl_xor(ss, o, 32); }
  float mean = s * (1.f / 256.f);
  float var  = ss * (1.f / 256.f) - mean * mean;
  float inv  = rsqrtf(var + 1e-5f);
  float* orow = out + (size_t)row * 256;
#pragma unroll
  for (int i = 0; i < 8; ++i) {
    int cidx = lid + i * 32;
    orow[cidx] = (v[i] - mean) * inv * g[cidx] + b[cidx];
  }
}

__global__ void pool_sum_bf16(const unsigned short* __restrict__ x,
                              const int* __restrict__ batch,
                              float* __restrict__ pool, int n) {
  int idx = blockIdx.x * blockDim.x + threadIdx.x;
  if (idx >= n * 256) return;
  int node = idx >> 8, col = idx & 255;
  atomicAdd(&pool[(size_t)batch[node] * 256 + col], bf16_to_f32(x[idx]));
}
__global__ void pool_count(const int* __restrict__ batch, float* cnt, int n) {
  int i = blockIdx.x * blockDim.x + threadIdx.x;
  if (i < n) atomicAdd(&cnt[batch[i]], 1.0f);
}
__global__ void pool_div(float* pool, const float* cnt) {
  int idx = blockIdx.x * blockDim.x + threadIdx.x;
  if (idx >= NBATCH * 256) return;
  pool[idx] /= fmaxf(cnt[idx >> 8], 1.0f);
}

// ---- small fp32 linear: out = act(concat(A,B) @ W + bias [+ res]) ---------
__global__ void linear_kernel(const float* __restrict__ inA, const float* __restrict__ inB,
                              int kA, int kB, const float* __restrict__ W,
                              const float* __restrict__ bias, const float* __restrict__ res,
                              float* __restrict__ out, int M, int Nout, int relu) {
  int idx = blockIdx.x * blockDim.x + threadIdx.x;
  if (idx >= M * Nout) return;
  int m = idx / Nout, n = idx % Nout;
  float acc = bias ? bias[n] : 0.f;
  const float* a = inA + (size_t)m * kA;
  for (int k = 0; k < kA; ++k) acc += a[k] * W[(size_t)k * Nout + n];
  if (inB) {
    const float* bb = inB + (size_t)m * kB;
    for (int k = 0; k < kB; ++k) acc += bb[k] * W[(size_t)(kA + k) * Nout + n];
  }
  if (res) acc += res[idx];
  if (relu) acc = fmaxf(acc, 0.f);
  out[idx] = acc;
}

__global__ void sigmoid_kernel(const float* __restrict__ logits, float* probs, int n) {
  int i = blockIdx.x * blockDim.x + threadIdx.x;
  if (i < n) probs[i] = 1.f / (1.f + expf(-logits[i]));
}

// ---------------------------------------------------------------------------
// d_in layout: 8 data tensors, then params flattened with sorted dict keys.
enum {
  IN_LIT_X = 0, IN_KG_X, IN_LIT_EI, IN_KG_EI, IN_LIT_EA, IN_KG_EA, IN_LIT_B, IN_KG_B,
  P_CLF1_W = 8, P_CLF1_B, P_CLF2_W, P_CLF2_B, P_CLF3_W, P_CLF3_B,
  P_FUS1_W, P_FUS1_B, P_FUS2_W, P_FUS2_B,
  P_K2L_LNB, P_K2L_LNG, P_K2L_WK_W, P_K2L_WK_B, P_K2L_WO_W, P_K2L_WO_B,
  P_K2L_WQ_W, P_K2L_WQ_B, P_K2L_WV_W, P_K2L_WV_B,
  P_KG_ENC0 = 28,                 // per layer: We, Wn, b, g, msg.W, msg.b (x3)
  P_KG_OUT_W = 46, P_KG_OUT_B, P_KG_PROJ_W, P_KG_PROJ_B,
  P_L2K_LNB = 50, P_L2K_LNG, P_L2K_WK_W, P_L2K_WK_B, P_L2K_WO_W, P_L2K_WO_B,
  P_L2K_WQ_W, P_L2K_WQ_B, P_L2K_WV_W, P_L2K_WV_B,
  P_LIT_ENC0 = 60,
  P_LIT_OUT_W = 78, P_LIT_OUT_B, P_LIT_PROJ_W, P_LIT_PROJ_B
};

extern "C" void kernel_launch(void* const* d_in, const int* in_sizes, int n_in,
                              void* d_out, int out_size, void* d_ws, size_t ws_size,
                              hipStream_t stream) {
  const int N_  = in_sizes[IN_LIT_B];        // 20000 nodes
  const int E_  = in_sizes[IN_LIT_EI] / 2;   // 160000 edges
  const int N64 = ((N_ + 63) / 64) * 64;     // padded rows for guard-free GEMM
  float* out = (float*)d_out;                // [logits 64 | probs 64 | lit_enh | kg_enh]

  // ---- workspace carve ----
  char* ws = (char*)d_ws;  size_t off = 0;
  auto carve = [&](size_t bytes) { off = (off + 255) & ~(size_t)255; void* p = ws + off; off += bytes; return p; };
  float*          agg     = (float*)carve((size_t)N64 * 256 * 4);
  unsigned short* xb      = (unsigned short*)carve((size_t)N64 * 256 * 2);  // bf16 activations
  unsigned short* xtb     = (unsigned short*)carve((size_t)N64 * 256 * 2);  // bf16 xt = x@Wn
  unsigned short* packWn  = (unsigned short*)carve(256 * 256 * 2);
  unsigned short* packWe  = (unsigned short*)carve(64 * 256 * 2);
  unsigned short* packMsg = (unsigned short*)carve(512 * 256 * 2);
  float*          poolL   = (float*)carve(NBATCH * 256 * 4);
  float*          poolK   = (float*)carve(NBATCH * 256 * 4);
  float*          cnt     = (float*)carve(NBATCH * 4);
  float*          lproj   = (float*)carve(NBATCH * 256 * 4);
  float*          kproj   = (float*)carve(NBATCH * 256 * 4);
  float*          tmpV    = (float*)carve(NBATCH * 256 * 4);
  float*          tmpO    = (float*)carve(NBATCH * 256 * 4);
  float*          attL    = (float*)carve(NBATCH * 256 * 4);
  float*          attK    = (float*)carve(NBATCH * 256 * 4);
  float*          fusH    = (float*)carve(NBATCH * 256 * 4);
  float*          fusO    = (float*)carve(NBATCH * 256 * 4);
  float*          clfA    = (float*)carve(NBATCH * 256 * 4);
  float*          clfBuf  = (float*)carve(NBATCH * 128 * 4);

  auto lin = [&](const float* A, const float* Bp, int kA, int kB, const float* W,
                 const float* bias, const float* res, float* o, int M, int Nout, int relu) {
    int n = M * Nout;
    linear_kernel<<<(n + 255) / 256, 256, 0, stream>>>(A, Bp, kA, kB, W, bias, res, o, M, Nout, relu);
  };

  // ---- graph encoder: 3 fused GConv layers + mean pool ----
  auto encoder = [&](const float* x0, const int* ei, const float* ea,
                     const int* batch, int pbase, float* pool) {
    cvt_f32_bf16<<<((N_ * 256 / 4) + 255) / 256, 256, 0, stream>>>(x0, xb, N_ * 256);
    if (N64 > N_) {   // zero pad rows once; LN never touches them afterwards
      int pad = (N64 - N_) * 256;
      zero_u16<<<(pad + 255) / 256, 256, 0, stream>>>(xb + (size_t)N_ * 256, pad);
    }
    for (int l = 0; l < 3; ++l) {
      const float* We  = (const float*)d_in[pbase + l * 6 + 0];
      const float* Wn  = (const float*)d_in[pbase + l * 6 + 1];
      const float* lnb = (const float*)d_in[pbase + l * 6 + 2];
      const float* lng = (const float*)d_in[pbase + l * 6 + 3];
      const float* mW  = (const float*)d_in[pbase + l * 6 + 4];
      const float* mB  = (const float*)d_in[pbase + l * 6 + 5];
      pack_w_bf16<<<(256 * 256) / 256, 256, 0, stream>>>(Wn, packWn, 256, 256);
      pack_w_bf16<<<(64 * 256) / 256, 256, 0, stream>>>(We, packWe, 64, 256);
      pack_w_bf16<<<(512 * 256) / 256, 256, 0, stream>>>(mW, packMsg, 512, 256);
      int tiles = (N64 / 64) * 16;
      gemm_xw_bf16<<<(tiles + 7) / 8, 256, 0, stream>>>(xb, packWn, xtb, N64, 256, 256);
      zero_f32<<<((N_ * 256) + 255) / 256, 256, 0, stream>>>(agg, N_ * 256);
      edge_msg_scatter<<<(E_ + 31) / 32, 256, 0, stream>>>(xtb, ei, ea, packMsg, mB, packWe, agg, E_);
      ln_relu_bf16<<<(N_ + 7) / 8, 256, 0, stream>>>(agg, lng, lnb, xb, N_);
    }
    zero_f32<<<(NBATCH * 256 + 255) / 256, 256, 0, stream>>>(pool, NBATCH * 256);
    zero_f32<<<1, 256, 0, stream>>>(cnt, NBATCH);
    pool_sum_bf16<<<((N_ * 256) + 255) / 256, 256, 0, stream>>>(xb, batch, pool, N_);
    pool_count<<<(N_ + 255) / 256, 256, 0, stream>>>(batch, cnt, N_);
    pool_div<<<(NBATCH * 256 + 255) / 256, 256, 0, stream>>>(pool, cnt);
  };

  encoder((const float*)d_in[IN_LIT_X], (const int*)d_in[IN_LIT_EI],
          (const float*)d_in[IN_LIT_EA], (const int*)d_in[IN_LIT_B], P_LIT_ENC0, poolL);
  encoder((const float*)d_in[IN_KG_X], (const int*)d_in[IN_KG_EI],
          (const float*)d_in[IN_KG_EA], (const int*)d_in[IN_KG_B], P_KG_ENC0, poolK);

  // ---- projections ----
  lin(poolL, nullptr, 256, 0, (const float*)d_in[P_LIT_PROJ_W], (const float*)d_in[P_LIT_PROJ_B],
      nullptr, lproj, NBATCH, 256, 0);
  lin(poolK, nullptr, 256, 0, (const float*)d_in[P_KG_PROJ_W], (const float*)d_in[P_KG_PROJ_B],
      nullptr, kproj, NBATCH, 256, 0);

  // ---- MHA, seq-len 1 => softmax over single key == 1 => out = wo(wv(k)) --
  lin(kproj, nullptr, 256, 0, (const float*)d_in[P_L2K_WV_W], (const float*)d_in[P_L2K_WV_B],
      nullptr, tmpV, NBATCH, 256, 0);
  lin(tmpV, nullptr, 256, 0, (const float*)d_in[P_L2K_WO_W], (const float*)d_in[P_L2K_WO_B],
      nullptr, tmpO, NBATCH, 256, 0);
  ln_kernel<<<(NBATCH + 7) / 8, 256, 0, stream>>>(tmpO, lproj,
      (const float*)d_in[P_L2K_LNG], (const float*)d_in[P_L2K_LNB], attL, NBATCH);
  lin(lproj, nullptr, 256, 0, (const float*)d_in[P_K2L_WV_W], (const float*)d_in[P_K2L_WV_B],
      nullptr, tmpV, NBATCH, 256, 0);
  lin(tmpV, nullptr, 256, 0, (const float*)d_in[P_K2L_WO_W], (const float*)d_in[P_K2L_WO_B],
      nullptr, tmpO, NBATCH, 256, 0);
  ln_kernel<<<(NBATCH + 7) / 8, 256, 0, stream>>>(tmpO, kproj,
      (const float*)d_in[P_K2L_LNG], (const float*)d_in[P_K2L_LNB], attK, NBATCH);

  // ---- fusion + outs (fus1/fus2 shared) ----
  float* litEnh = out + 128;
  float* kgEnh  = out + 128 + NBATCH * 256;
  lin(lproj, attL, 256, 256, (const float*)d_in[P_FUS1_W], (const float*)d_in[P_FUS1_B],
      nullptr, fusH, NBATCH, 256, 1);
  lin(fusH, nullptr, 256, 0, (const float*)d_in[P_FUS2_W], (const float*)d_in[P_FUS2_B],
      nullptr, fusO, NBATCH, 256, 0);
  lin(fusO, nullptr, 256, 0, (const float*)d_in[P_LIT_OUT_W], (const float*)d_in[P_LIT_OUT_B],
      poolL, litEnh, NBATCH, 256, 0);
  lin(kproj, attK, 256, 256, (const float*)d_in[P_FUS1_W], (const float*)d_in[P_FUS1_B],
      nullptr, fusH, NBATCH, 256, 1);
  lin(fusH, nullptr, 256, 0, (const float*)d_in[P_FUS2_W], (const float*)d_in[P_FUS2_B],
      nullptr, fusO, NBATCH, 256, 0);
  lin(fusO, nullptr, 256, 0, (const float*)d_in[P_KG_OUT_W], (const float*)d_in[P_KG_OUT_B],
      poolK, kgEnh, NBATCH, 256, 0);

  // ---- classifier ----
  lin(litEnh, kgEnh, 256, 256, (const float*)d_in[P_CLF1_W], (const float*)d_in[P_CLF1_B],
      nullptr, clfA, NBATCH, 256, 1);
  lin(clfA, nullptr, 256, 0, (const float*)d_in[P_CLF2_W], (const float*)d_in[P_CLF2_B],
      nullptr, clfBuf, NBATCH, 128, 1);
  lin(clfBuf, nullptr, 128, 0, (const float*)d_in[P_CLF3_W], (const float*)d_in[P_CLF3_B],
      nullptr, out, NBATCH, 1, 0);
  sigmoid_kernel<<<1, 256, 0, stream>>>(out, out + 64, NBATCH);
  (void)n_in; (void)out_size; (void)ws_size;
}
